// Attention_21981642621194
// MI455X (gfx1250) — matrix operand order repaired and checked by reading, not compiled, not run on hardware
//
#include <hip/hip_runtime.h>
#include <hip/hip_bf16.h>

typedef __attribute__((ext_vector_type(16))) __bf16 v16bf;
typedef __attribute__((ext_vector_type(8)))  float  v8f;

constexpr int kB = 4;
constexpr int kS = 4096;
constexpr int kE = 512;
constexpr int QT = 32;   // query rows per workgroup
constexpr int KT = 64;   // keys per flash iteration

struct U4x2 { uint4 a, b; };

__device__ __forceinline__ unsigned short f2bf(float f) {
  unsigned u = __builtin_bit_cast(unsigned, f);
  u += 0x7FFFu + ((u >> 16) & 1u);          // round-to-nearest-even
  return (unsigned short)(u >> 16);
}

// A-operand (16x32 bf16): lane<16 holds K[k..k+7],K[k+16..k+23]; lane>=16 the other 8+8
__device__ __forceinline__ v16bf fragA(const unsigned short* rowk, int half) {
  U4x2 t;
  t.a = *reinterpret_cast<const uint4*>(rowk + half * 8);
  t.b = *reinterpret_cast<const uint4*>(rowk + 16 + half * 8);
  return __builtin_bit_cast(v16bf, t);
}
// B-operand (32x16 bf16): lane<16 holds K[k..k+7],K[k+16..k+23] of its column; lane>=16 the other 8+8
__device__ __forceinline__ v16bf fragB(const unsigned short* colk, int half) {
  U4x2 t;
  t.a = *reinterpret_cast<const uint4*>(colk + half * 8);
  t.b = *reinterpret_cast<const uint4*>(colk + 16 + half * 8);
  return __builtin_bit_cast(v16bf, t);
}

__device__ __forceinline__ float redmax16(float v) {
  v = fmaxf(v, __shfl_xor(v, 1));
  v = fmaxf(v, __shfl_xor(v, 2));
  v = fmaxf(v, __shfl_xor(v, 4));
  v = fmaxf(v, __shfl_xor(v, 8));
  return v;
}
__device__ __forceinline__ float redsum16(float v) {
  v += __shfl_xor(v, 1);
  v += __shfl_xor(v, 2);
  v += __shfl_xor(v, 4);
  v += __shfl_xor(v, 8);
  return v;
}

// ---------------------------------------------------------------- convert
__global__ void __launch_bounds__(256) cvt_bf16(const float* __restrict__ src,
                                                unsigned short* __restrict__ dst, int n) {
  int i = blockIdx.x * 256 + threadIdx.x;
  if (i < n) dst[i] = f2bf(src[i]);
}

// ---------------------------------------------------------------- QKV GEMM
// Y = x @ W^T + bias ; sel 0->Q (row major), 1->K (row major), 2->V stored transposed
__global__ void __launch_bounds__(256) qkv_gemm(
    const unsigned short* __restrict__ xb, const unsigned short* __restrict__ Wb,
    const float* __restrict__ bq, const float* __restrict__ bk, const float* __restrict__ bvv,
    unsigned short* __restrict__ Qb, unsigned short* __restrict__ Kb,
    unsigned short* __restrict__ Vt) {
  const int sel = blockIdx.z;
  const unsigned short* W = Wb + sel * kE * kE;
  const float* bias = (sel == 0) ? bq : (sel == 1) ? bk : bvv;

  const int tid = threadIdx.x, wave = tid >> 5, lane = tid & 31;
  const int half = lane >> 4, lr = lane & 15;
  const int ms = blockIdx.x * 128 + wave * 16;   // 16-row strip per wave
  const int nb = blockIdx.y * 64;                // 64 output columns per block

  v8f acc[4];
#pragma unroll
  for (int t = 0; t < 4; t++)
#pragma unroll
    for (int v = 0; v < 8; v++) acc[t][v] = 0.f;

  const unsigned short* arow = xb + (size_t)(ms + lr) * kE;
#pragma unroll 4
  for (int k = 0; k < kE; k += 32) {
    v16bf a = fragA(arow + k, half);
#pragma unroll
    for (int nt = 0; nt < 4; nt++) {
      const unsigned short* brow = W + (size_t)(nb + nt * 16 + lr) * kE;
      v16bf b = fragB(brow + k, half);
      acc[nt] = __builtin_amdgcn_wmma_f32_16x16x32_bf16(false, a, false, b,
                                                        (short)0, acc[nt], false, false);
    }
  }

#pragma unroll
  for (int nt = 0; nt < 4; nt++) {
    const int n = nb + nt * 16 + lr;
    const float bn = bias[n];
#pragma unroll
    for (int v = 0; v < 8; v++) {
      const int m = ms + v + 8 * half;           // D layout: row = vgpr + 8*(lane>=16)
      const unsigned short h = f2bf(acc[nt][v] + bn);
      if (sel == 0)       Qb[(size_t)m * kE + n] = h;
      else if (sel == 1)  Kb[(size_t)m * kE + n] = h;
      else {
        const int bi = m >> 12, s = m & (kS - 1);
        Vt[((size_t)(bi * kE + n) << 12) + s] = h;   // Vt[b][e][s]
      }
    }
  }
}

// ---------------------------------------------------------------- flash attention + LN
__global__ void __launch_bounds__(256) attn_ln(
    const unsigned short* __restrict__ Qb, const unsigned short* __restrict__ Kb,
    const unsigned short* __restrict__ Vt, const int* __restrict__ mask,
    const float* __restrict__ x, const float* __restrict__ gamma,
    const float* __restrict__ beta, float* __restrict__ out) {
  __shared__ unsigned short Qs[QT * kE];  // 32 KB query tile (bf16)
  __shared__ unsigned short Ps[QT * KT];  // 4 KB probability tile (bf16)
  __shared__ float redMax[QT * 4];
  __shared__ float redSum[QT * 4];
  __shared__ float rowM[QT], rowL[QT], rowFac[QT];
  __shared__ float redS[QT * 8], redQ[QT * 8];
  __shared__ float rowMu[QT], rowRs[QT];

  const int b = blockIdx.y;
  const int qb0 = blockIdx.x * QT;
  const int tid = threadIdx.x, wave = tid >> 5, lane = tid & 31;
  const int half = lane >> 4, lr = lane & 15;
  const int mi = wave >> 2, ni = wave & 3;      // score-tile assignment (2x4 tiles of 16x16)

  {  // stage contiguous 32 KB Q tile into LDS with CDNA5 async-to-LDS DMA
    const uint4* src = reinterpret_cast<const uint4*>(Qb + (size_t)(b * kS + qb0) * kE);
    const unsigned lds_base = (unsigned)(uintptr_t)(&Qs[0]);  // low 32 bits = LDS offset
#pragma unroll
    for (int i = 0; i < (QT * kE / 8) / 256; i++) {
      const int idx = tid + i * 256;                 // uint4 index
      const uint4* g = src + idx;
      unsigned lds = lds_base + idx * 16;
      asm volatile("global_load_async_to_lds_b128 %0, %1, off"
                   :: "v"(lds), "v"(g) : "memory");
    }
    asm volatile("s_wait_asynccnt 0" ::: "memory");
  }
  if (tid < QT) { rowM[tid] = -1e30f; rowL[tid] = 0.f; }
  __syncthreads();

  v8f ctx[2][4];   // 32 rows x 64 e-cols per wave
#pragma unroll
  for (int a = 0; a < 2; a++)
#pragma unroll
    for (int c = 0; c < 4; c++)
#pragma unroll
      for (int v = 0; v < 8; v++) ctx[a][c][v] = 0.f;

  const float scale = 0.044194173824159216f;   // 1/sqrt(512)

  for (int kb = 0; kb < kS; kb += KT) {
    // prefetch next K/V blocks into cache (streaming operands, L2-latency hiding)
    if (kb + KT < kS) {
      const unsigned short* nk =
          Kb + (size_t)(b * kS + kb + KT + ni * 16 + lr) * kE;
      __builtin_prefetch(nk + half * 256, 0, 1);
      const unsigned short* nv =
          Vt + (size_t)(b * kE + wave * 64 + half * 32 + lr) * kS + kb + KT;
      __builtin_prefetch(nv, 0, 1);
      __builtin_prefetch(nv + (size_t)16 * kS, 0, 1);
    }
    // -------- scores: S_tile(32x64) = Q(32x512) @ K_block^T
    v8f sc;
#pragma unroll
    for (int v = 0; v < 8; v++) sc[v] = 0.f;
    const unsigned short* arow = Qs + (mi * 16 + lr) * kE;
    const unsigned short* brow = Kb + (size_t)(b * kS + kb + ni * 16 + lr) * kE;
#pragma unroll 4
    for (int k = 0; k < kE; k += 32) {
      v16bf a = fragA(arow + k, half);
      v16bf bm = fragB(brow + k, half);
      sc = __builtin_amdgcn_wmma_f32_16x16x32_bf16(false, a, false, bm,
                                                   (short)0, sc, false, false);
    }
    const int col = kb + ni * 16 + lr;
    const int mk = mask[b * kS + col];
    float rmax[8];
#pragma unroll
    for (int v = 0; v < 8; v++) {
      float t = mk ? sc[v] * scale : -1e9f;
      sc[v] = t;
      rmax[v] = redmax16(t);
    }
    if (lr == 0) {
#pragma unroll
      for (int v = 0; v < 8; v++) redMax[(mi * 16 + v + 8 * half) * 4 + ni] = rmax[v];
    }
    __syncthreads();
    if (tid < QT) {
      float old = rowM[tid], nm = old;
#pragma unroll
      for (int j = 0; j < 4; j++) nm = fmaxf(nm, redMax[tid * 4 + j]);
      rowM[tid] = nm;
      rowFac[tid] = __expf(old - nm);
    }
    __syncthreads();
    // -------- P = exp(S - m); block row-sums; P -> LDS (bf16)
#pragma unroll
    for (int v = 0; v < 8; v++) {
      const int row = mi * 16 + v + 8 * half;
      float p = __expf(sc[v] - rowM[row]);
      Ps[row * KT + ni * 16 + lr] = f2bf(p);
      float s = redsum16(p);
      if (lr == 0) redSum[row * 4 + ni] = s;
    }
    __syncthreads();
    if (tid < QT) {
      float bs = redSum[tid * 4 + 0] + redSum[tid * 4 + 1] +
                 redSum[tid * 4 + 2] + redSum[tid * 4 + 3];
      rowL[tid] = rowL[tid] * rowFac[tid] + bs;
    }
    // -------- context: rescale accumulators, += P(32x64) @ V(64x64-chunk)
#pragma unroll
    for (int m2 = 0; m2 < 2; m2++) {
      float fac[8];
#pragma unroll
      for (int v = 0; v < 8; v++) fac[v] = rowFac[m2 * 16 + v + 8 * half];
      const unsigned short* prow = Ps + (m2 * 16 + lr) * KT;
#pragma unroll
      for (int e2 = 0; e2 < 4; e2++) {
        v8f c = ctx[m2][e2];
#pragma unroll
        for (int v = 0; v < 8; v++) c[v] *= fac[v];
        const unsigned short* vrow =
            Vt + (size_t)(b * kE + wave * 64 + e2 * 16 + lr) * kS + kb;
#pragma unroll
        for (int k2 = 0; k2 < KT; k2 += 32) {
          v16bf a = fragA(prow + k2, half);
          v16bf bm = fragB(vrow + k2, half);
          c = __builtin_amdgcn_wmma_f32_16x16x32_bf16(false, a, false, bm,
                                                      (short)0, c, false, false);
        }
        ctx[m2][e2] = c;
      }
    }
    __syncthreads();
  }

  // -------- epilogue: h = ctx/l + x ; LayerNorm(h)*gamma + beta
#pragma unroll
  for (int m2 = 0; m2 < 2; m2++) {
    float il[8];
#pragma unroll
    for (int v = 0; v < 8; v++) il[v] = 1.f / rowL[m2 * 16 + v + 8 * half];
#pragma unroll
    for (int e2 = 0; e2 < 4; e2++) {
      const int colc = wave * 64 + e2 * 16 + lr;
#pragma unroll
      for (int v = 0; v < 8; v++) {
        const int row = qb0 + m2 * 16 + v + 8 * half;
        ctx[m2][e2][v] = ctx[m2][e2][v] * il[v] + x[(size_t)(b * kS + row) * kE + colc];
      }
    }
#pragma unroll
    for (int v = 0; v < 8; v++) {
      float s = 0.f, q = 0.f;
#pragma unroll
      for (int e2 = 0; e2 < 4; e2++) { float h = ctx[m2][e2][v]; s += h; q += h * h; }
      s = redsum16(s); q = redsum16(q);
      if (lr == 0) {
        const int row = m2 * 16 + v + 8 * half;
        redS[row * 8 + wave] = s; redQ[row * 8 + wave] = q;
      }
    }
  }
  __syncthreads();
  if (tid < QT) {
    float s = 0.f, q = 0.f;
#pragma unroll
    for (int w = 0; w < 8; w++) { s += redS[tid * 8 + w]; q += redQ[tid * 8 + w]; }
    float mu = s * (1.0f / kE);
    float var = q * (1.0f / kE) - mu * mu;
    rowMu[tid] = mu; rowRs[tid] = rsqrtf(var + 1e-5f);
  }
  __syncthreads();
#pragma unroll
  for (int m2 = 0; m2 < 2; m2++) {
#pragma unroll
    for (int e2 = 0; e2 < 4; e2++) {
      const int colc = wave * 64 + e2 * 16 + lr;
      const float g = gamma[colc], be = beta[colc];
#pragma unroll
      for (int v = 0; v < 8; v++) {
        const int rt = m2 * 16 + v + 8 * half;
        out[(size_t)(b * kS + qb0 + rt) * kE + colc] =
            (ctx[m2][e2][v] - rowMu[rt]) * rowRs[rt] * g + be;
      }
    }
  }
}

// ---------------------------------------------------------------- launch
extern "C" void kernel_launch(void* const* d_in, const int* in_sizes, int n_in,
                              void* d_out, int out_size, void* d_ws, size_t ws_size,
                              hipStream_t stream) {
  (void)in_sizes; (void)n_in; (void)out_size; (void)ws_size;
  const float* x     = (const float*)d_in[0];
  const int*   mask  = (const int*)d_in[1];
  const float* Wq    = (const float*)d_in[2];
  const float* bq    = (const float*)d_in[3];
  const float* Wk    = (const float*)d_in[4];
  const float* bk    = (const float*)d_in[5];
  const float* Wv    = (const float*)d_in[6];
  const float* bv    = (const float*)d_in[7];
  const float* gamma = (const float*)d_in[8];
  const float* beta  = (const float*)d_in[9];
  float* out = (float*)d_out;

  // workspace layout (bytes): xb 16MB | Wb 1.5MB | Qb 16MB | Kb 16MB | Vt 16MB
  char* ws = (char*)d_ws;
  unsigned short* xb = (unsigned short*)(ws);
  unsigned short* Wb = (unsigned short*)(ws + 16777216);
  unsigned short* Qb = (unsigned short*)(ws + 18350080);
  unsigned short* Kb = (unsigned short*)(ws + 35127296);
  unsigned short* Vt = (unsigned short*)(ws + 51904512);

  const int nx = kB * kS * kE;
  const int nw = kE * kE;
  cvt_bf16<<<dim3((nx + 255) / 256), 256, 0, stream>>>(x, xb, nx);
  cvt_bf16<<<dim3((nw + 255) / 256), 256, 0, stream>>>(Wq, Wb, nw);
  cvt_bf16<<<dim3((nw + 255) / 256), 256, 0, stream>>>(Wk, Wb + nw, nw);
  cvt_bf16<<<dim3((nw + 255) / 256), 256, 0, stream>>>(Wv, Wb + 2 * nw, nw);

  qkv_gemm<<<dim3(kB * kS / 128, kE / 64, 3), 256, 0, stream>>>(
      xb, Wb, bq, bk, bv, Qb, Kb, Vt);

  attn_ln<<<dim3(kS / QT, kB), 256, 0, stream>>>(
      Qb, Kb, Vt, mask, x, gamma, beta, out);
}
